// CausalSelfAttention_61727269978186
// MI455X (gfx1250) — matrix operand order repaired and checked by reading, not compiled, not run on hardware
//
#include <hip/hip_runtime.h>
#include <hip/hip_bf16.h>
#include <stdint.h>

// ---------------------------------------------------------------------------
// CDNA5 (gfx1250, wave32) causal self-attention with nGPT-style cosine-norm +
// rotary, built on v_wmma_f32_16x16x32_bf16. 5 launches total.
// ---------------------------------------------------------------------------

typedef __bf16 bf16;
typedef __attribute__((ext_vector_type(16))) __bf16 v16bf;
typedef __attribute__((ext_vector_type(8)))  __bf16 v8bf;
typedef __attribute__((ext_vector_type(8)))  float   v8f;

// GCC-style int4 (matches the async-LDS builtin's pointee type) + AS variants.
typedef int v4i_t __attribute__((vector_size(16)));
typedef __attribute__((address_space(1))) v4i_t gv4i;
typedef __attribute__((address_space(3))) v4i_t lv4i;

#define T_SEQ   2048
#define N_HEADS 16
#define HD      64
#define DMODEL  1024
#define SCALE_ATTN 0.12f

#if defined(__HIP_DEVICE_COMPILE__) && __has_builtin(__builtin_amdgcn_global_load_async_to_lds_b128)
#define HAVE_ASYNC_LDS 1
#else
#define HAVE_ASYNC_LDS 0
#endif

// --- WMMA wrapper (8-arg bf16 form, probe-confirmed signature) --------------
static __device__ __forceinline__ v8f wmma_bf16(v16bf a, v16bf b, v8f c) {
  return __builtin_amdgcn_wmma_f32_16x16x32_bf16(false, a, false, b,
                                                 (short)0, c, false, false);
}

// Assemble a 16-element bf16 fragment from two 16-byte chunks.
union V16U { v16bf v; v8bf h8[2]; };
static __device__ __forceinline__ v16bf ld16(const bf16* lo, const bf16* hi) {
  V16U u;
  u.h8[0] = *(const v8bf*)lo;
  u.h8[1] = *(const v8bf*)hi;
  return u.v;
}

// --- wave32 reductions ------------------------------------------------------
static __device__ __forceinline__ float redmax16(float x) {
  x = fmaxf(x, __shfl_xor(x, 1, 32));
  x = fmaxf(x, __shfl_xor(x, 2, 32));
  x = fmaxf(x, __shfl_xor(x, 4, 32));
  x = fmaxf(x, __shfl_xor(x, 8, 32));
  return x;
}
static __device__ __forceinline__ float redsum16(float x) {
  x += __shfl_xor(x, 1, 32);
  x += __shfl_xor(x, 2, 32);
  x += __shfl_xor(x, 4, 32);
  x += __shfl_xor(x, 8, 32);
  return x;
}
static __device__ __forceinline__ float redsum32(float x) {
  x += __shfl_xor(x, 1, 32);
  x += __shfl_xor(x, 2, 32);
  x += __shfl_xor(x, 4, 32);
  x += __shfl_xor(x, 8, 32);
  x += __shfl_xor(x, 16, 32);
  return x;
}

// --- fused f32 -> bf16 conversion for x + 4 weights (one launch) ------------
// Blocks [0,2048): x (2M elems); then 4 x 1024 blocks for Wq/Wk/Wv/Wo (1M each).
__global__ __launch_bounds__(256) void cvt_all_kernel(
    const float* __restrict__ x,  const float* __restrict__ wq,
    const float* __restrict__ wk, const float* __restrict__ wv,
    const float* __restrict__ wo,
    bf16* __restrict__ xb,  bf16* __restrict__ wqb, bf16* __restrict__ wkb,
    bf16* __restrict__ wvb, bf16* __restrict__ wob) {
  const int b = blockIdx.x;
  const float* src;
  bf16* dst;
  int blk;
  if (b < 2048) { src = x; dst = xb; blk = b; }
  else {
    const int w  = (b - 2048) >> 10;   // 0..3
    blk = (b - 2048) & 1023;
    src = (w == 0) ? wq : (w == 1) ? wk : (w == 2) ? wv : wo;
    dst = (w == 0) ? wqb : (w == 1) ? wkb : (w == 2) ? wvb : wob;
  }
  const int i = blk * 256 + threadIdx.x;   // index in float4 units
  float4 f = ((const float4*)src)[i];
  const int o = i * 4;
  dst[o + 0] = (bf16)f.x;
  dst[o + 1] = (bf16)f.y;
  dst[o + 2] = (bf16)f.z;
  dst[o + 3] = (bf16)f.w;
}

// --- GEMM body: C[2048x1024] f32 = A[2048x1024] bf16 @ B[1024x1024] bf16 ----
// Block 256 threads (8 waves), tile 128(M) x 64(N), K-step 32.
// A staged in LDS (async global->LDS when available), B staged transposed so
// B fragments are contiguous 32B DS reads.
static __device__ __forceinline__ void gemm_body(const bf16* __restrict__ A,
                                                 const bf16* __restrict__ B,
                                                 float* __restrict__ C) {
  __shared__ bf16 As[128 * 40];   // 10240 B
  __shared__ bf16 Bt[64 * 40];    //  5120 B
  const int tid  = threadIdx.x;
  const int lane = tid & 31;
  const int hf   = lane >> 4;     // wave half (0/1)
  const int nl   = lane & 15;
  const int wave = tid >> 5;
  const int wm   = (wave & 3) * 32;
  const int wn   = (wave >> 2) * 32;
  const int bm   = blockIdx.x * 128;
  const int bn   = blockIdx.y * 64;
  const int arow = tid >> 1, acol = (tid & 1) * 16;  // A stage: 16 bf16/thread
  const int brow = tid >> 3, bcol = (tid & 7) * 8;   // B stage:  8 bf16/thread

  v8f acc[2][2] = {};

  for (int k0 = 0; k0 < DMODEL; k0 += 32) {
    const bf16* agp = A + (size_t)(bm + arow) * DMODEL + k0 + acol;
    const uint4 bv0 = *(const uint4*)(B + (size_t)(k0 + brow) * DMODEL + bn + bcol);
#if !HAVE_ASYNC_LDS
    const uint4 av0 = ((const uint4*)agp)[0];
    const uint4 av1 = ((const uint4*)agp)[1];
#endif
    __syncthreads();
#if HAVE_ASYNC_LDS
    {
      bf16* lp = As + arow * 40 + acol;
      __builtin_amdgcn_global_load_async_to_lds_b128((gv4i*)agp, (lv4i*)lp, 0, 0);
      __builtin_amdgcn_global_load_async_to_lds_b128((gv4i*)(agp + 8),
                                                     (lv4i*)(lp + 8), 0, 0);
    }
#else
    *(uint4*)(As + arow * 40 + acol)     = av0;
    *(uint4*)(As + arow * 40 + acol + 8) = av1;
#endif
    const bf16* bl = (const bf16*)&bv0;
#pragma unroll
    for (int jj = 0; jj < 8; ++jj) Bt[(bcol + jj) * 40 + brow] = bl[jj];
#if HAVE_ASYNC_LDS
    asm volatile("s_wait_asynccnt 0" ::: "memory");
#endif
    __syncthreads();

    v16bf af[2], bfr[2];
#pragma unroll
    for (int mi = 0; mi < 2; ++mi) {           // A frag: k = hf*16 + i
      const bf16* p = As + (wm + mi * 16 + nl) * 40;
      af[mi] = ld16(p + hf * 16, p + hf * 16 + 8);
    }
#pragma unroll
    for (int ni = 0; ni < 2; ++ni) {           // B frag: k = hf*16 + i
      const bf16* p = Bt + (wn + ni * 16 + nl) * 40 + hf * 16;
      bfr[ni] = ld16(p, p + 8);
    }
#pragma unroll
    for (int mi = 0; mi < 2; ++mi)
#pragma unroll
      for (int ni = 0; ni < 2; ++ni)
        acc[mi][ni] = wmma_bf16(af[mi], bfr[ni], acc[mi][ni]);
  }

#pragma unroll
  for (int mi = 0; mi < 2; ++mi)
#pragma unroll
    for (int ni = 0; ni < 2; ++ni)
#pragma unroll
      for (int r = 0; r < 8; ++r)              // C/D layout: row r+hf*8, col nl
        C[(size_t)(bm + wm + mi * 16 + r + hf * 8) * DMODEL + bn + wn + ni * 16 + nl] =
            acc[mi][ni][r];
}

// Fused QKV projection: grid.z selects weight/output (one launch for 3 GEMMs).
__global__ __launch_bounds__(256) void gemm_qkv_kernel(
    const bf16* __restrict__ A,
    const bf16* __restrict__ Bq, const bf16* __restrict__ Bk,
    const bf16* __restrict__ Bv,
    float* __restrict__ Cq, float* __restrict__ Ck, float* __restrict__ Cv) {
  const int z = blockIdx.z;
  const bf16* B = (z == 0) ? Bq : (z == 1) ? Bk : Bv;
  float*      C = (z == 0) ? Cq : (z == 1) ? Ck : Cv;
  gemm_body(A, B, C);
}

__global__ __launch_bounds__(256) void gemm_single_kernel(const bf16* __restrict__ A,
                                                          const bf16* __restrict__ B,
                                                          float* __restrict__ C) {
  gemm_body(A, B, C);
}

// --- cosine-norm + s_qk scale + rotary; pack head-major bf16 ----------------
// One wave per (t, head). Lane d owns rotary pair (d, d+32).
__global__ __launch_bounds__(32) void normrot_kernel(const float* __restrict__ qf,
                                                     const float* __restrict__ kf,
                                                     const float* __restrict__ vf,
                                                     const float* __restrict__ sqk,
                                                     bf16* __restrict__ qh,
                                                     bf16* __restrict__ kh,
                                                     bf16* __restrict__ vt) {
  const int t = blockIdx.x;
  const int h = blockIdx.y;
  const int d = threadIdx.x;  // 0..31
  const size_t base = (size_t)t * DMODEL + (size_t)h * HD;
  const float se1 = sqk[h * HD + d] * 32.0f;        // sqrt(DIM)=32
  const float se2 = sqk[h * HD + d + 32] * 32.0f;
  // angular_freq[j] = (1/1024)^(j/15) for j<16 else 0
  const float fr  = (d < 16) ? exp2f(-10.0f * (float)d * (1.0f / 15.0f)) : 0.0f;
  const float th  = (float)t * fr;
  const float cth = cosf(th), sth = sinf(th);
  const size_t ob = ((size_t)h * T_SEQ + t) * HD;

  {
    float x1 = qf[base + d], x2 = qf[base + d + 32];
    float inv = rsqrtf(redsum32(x1 * x1 + x2 * x2) + 1e-12f);
    float a = x1 * inv * se1, b = x2 * inv * se2;
    qh[ob + d]      = (bf16)( a * cth + b * sth);
    qh[ob + d + 32] = (bf16)(-a * sth + b * cth);
  }
  {
    float x1 = kf[base + d], x2 = kf[base + d + 32];
    float inv = rsqrtf(redsum32(x1 * x1 + x2 * x2) + 1e-12f);
    float a = x1 * inv * se1, b = x2 * inv * se2;
    kh[ob + d]      = (bf16)( a * cth + b * sth);
    kh[ob + d + 32] = (bf16)(-a * sth + b * cth);
  }
  {  // V stored transposed [h][64][T] so PV B-fragments are contiguous
    vt[((size_t)h * HD + d) * T_SEQ + t]      = (bf16)vf[base + d];
    vt[((size_t)h * HD + d + 32) * T_SEQ + t] = (bf16)vf[base + d + 32];
  }
}

// --- flash attention: one wave per (16-query tile, head) --------------------
__global__ __launch_bounds__(32) void attn_kernel(const bf16* __restrict__ qh,
                                                  const bf16* __restrict__ kh,
                                                  const bf16* __restrict__ vt,
                                                  bf16* __restrict__ ybf) {
  __shared__ bf16 Pl[16 * 40];   // P tile staging (pad stride 40)
  const int qt   = blockIdx.x;   // 0..127
  const int h    = blockIdx.y;   // 0..15
  const int lane = threadIdx.x;
  const int hf   = lane >> 4;
  const int nl   = lane & 15;
  const int q0   = qt * 16;

  // Q fragments (reused every chunk): row m = nl, k = hf*16 + i (contiguous)
  const bf16* Qrow = qh + ((size_t)h * T_SEQ + q0 + nl) * HD;
  const v16bf aq0 = ld16(Qrow + hf * 16,      Qrow + hf * 16 + 8);   // d 0..31
  const v16bf aq1 = ld16(Qrow + 32 + hf * 16, Qrow + 32 + hf * 16 + 8);   // d 32..63

  float mrow[8], lrow[8];
  v8f o[4] = {};
#pragma unroll
  for (int r = 0; r < 8; ++r) { mrow[r] = -3.0e38f; lrow[r] = 0.0f; }

  const int nCh = (qt + 2) >> 1;           // ceil(16*(qt+1)/32)
  for (int j = 0; j < nCh; ++j) {
    const int kbase = j * 32;
    const bf16* Ka = kh + ((size_t)h * T_SEQ + kbase + nl) * HD;  // keys kbase+nl
    const bf16* Kb = Ka + 16 * HD;                                 // keys +16
    if (j + 1 < nCh) {  // prefetch next chunk's K/V stream (global_prefetch_b8)
      __builtin_prefetch(Ka + 32 * HD, 0, 1);
      __builtin_prefetch(Kb + 32 * HD, 0, 1);
      __builtin_prefetch(vt + ((size_t)h * HD + nl) * T_SEQ + kbase + 32, 0, 1);
    }
    v16bf bA0 = ld16(Ka + hf * 16,      Ka + hf * 16 + 8);
    v16bf bA1 = ld16(Ka + 32 + hf * 16, Ka + 32 + hf * 16 + 8);
    v16bf bB0 = ld16(Kb + hf * 16,      Kb + hf * 16 + 8);
    v16bf bB1 = ld16(Kb + 32 + hf * 16, Kb + 32 + hf * 16 + 8);

    v8f s0 = {}; s0 = wmma_bf16(aq0, bA0, s0); s0 = wmma_bf16(aq1, bA1, s0);
    v8f s1 = {}; s1 = wmma_bf16(aq0, bB0, s1); s1 = wmma_bf16(aq1, bB1, s1);

    const bool needMask = (kbase + 31 > q0);
    const int keyA = kbase + nl, keyB = keyA + 16;
#pragma unroll
    for (int r = 0; r < 8; ++r) {
      const int qrow = q0 + r + hf * 8;
      float v0 = s0[r] * SCALE_ATTN;
      float v1 = s1[r] * SCALE_ATTN;
      if (needMask) {
        if (keyA > qrow) v0 = -3.0e38f;
        if (keyB > qrow) v1 = -3.0e38f;
      }
      float mnew = fmaxf(mrow[r], redmax16(fmaxf(v0, v1)));
      float p0 = __expf(v0 - mnew);
      float p1 = __expf(v1 - mnew);
      float cf = __expf(mrow[r] - mnew);
      lrow[r] = lrow[r] * cf + redsum16(p0 + p1);
      mrow[r] = mnew;
      o[0][r] *= cf; o[1][r] *= cf; o[2][r] *= cf; o[3][r] *= cf;
      Pl[(r + hf * 8) * 40 + nl]      = (bf16)p0;   // C-layout -> LDS
      Pl[(r + hf * 8) * 40 + 16 + nl] = (bf16)p1;
    }
    asm volatile("" ::: "memory");  // per-wave DS ops are in-order; block reordering
    const bf16* Pr = Pl + nl * 40;
    v16bf ap = ld16(Pr + hf * 16, Pr + hf * 16 + 8);  // P as A-fragment (K=32 keys)
#pragma unroll
    for (int dt = 0; dt < 4; ++dt) {
      const bf16* Vp = vt + ((size_t)h * HD + dt * 16 + nl) * T_SEQ + kbase + hf * 16;
      v16bf bv = ld16(Vp, Vp + 8);
      o[dt] = wmma_bf16(ap, bv, o[dt]);
    }
    asm volatile("" ::: "memory");
  }

#pragma unroll
  for (int dt = 0; dt < 4; ++dt)
#pragma unroll
    for (int r = 0; r < 8; ++r) {
      float val = o[dt][r] / lrow[r];
      ybf[(size_t)(q0 + r + hf * 8) * DMODEL + h * HD + dt * 16 + nl] = (bf16)val;
    }
}

// ---------------------------------------------------------------------------
extern "C" void kernel_launch(void* const* d_in, const int* in_sizes, int n_in,
                              void* d_out, int out_size, void* d_ws, size_t ws_size,
                              hipStream_t stream) {
  (void)in_sizes; (void)n_in; (void)out_size; (void)ws_size;
  const float* x   = (const float*)d_in[0];
  const float* Wq  = (const float*)d_in[1];
  const float* Wk  = (const float*)d_in[2];
  const float* Wv  = (const float*)d_in[3];
  const float* Wo  = (const float*)d_in[4];
  const float* sqk = (const float*)d_in[5];
  float* out = (float*)d_out;
  char* ws = (char*)d_ws;

  const size_t MB = 1024ull * 1024ull;   // total workspace use: 52 MiB
  bf16*  x_bf  = (bf16*)(ws + 0 * MB);
  bf16*  wq_bf = (bf16*)(ws + 4 * MB);
  bf16*  wk_bf = (bf16*)(ws + 6 * MB);
  bf16*  wv_bf = (bf16*)(ws + 8 * MB);
  bf16*  wo_bf = (bf16*)(ws + 10 * MB);
  float* qf    = (float*)(ws + 12 * MB);
  float* kf    = (float*)(ws + 20 * MB);
  float* vf    = (float*)(ws + 28 * MB);
  bf16*  qhb   = (bf16*)(ws + 36 * MB);
  bf16*  khb   = (bf16*)(ws + 40 * MB);
  bf16*  vtb   = (bf16*)(ws + 44 * MB);
  bf16*  ybf   = (bf16*)(ws + 48 * MB);

  // 1) all f32->bf16 conversions in one launch
  cvt_all_kernel<<<6144, 256, 0, stream>>>(x, Wq, Wk, Wv, Wo,
                                           x_bf, wq_bf, wk_bf, wv_bf, wo_bf);
  // 2) fused QKV projection (grid.z = 3)
  gemm_qkv_kernel<<<dim3(16, 16, 3), 256, 0, stream>>>(x_bf, wq_bf, wk_bf, wv_bf,
                                                       qf, kf, vf);
  // 3) cosine-norm + rotary + head-major packing
  normrot_kernel<<<dim3(2048, 16), 32, 0, stream>>>(qf, kf, vf, sqk, qhb, khb, vtb);
  // 4) causal flash attention
  attn_kernel<<<dim3(128, 16), 32, 0, stream>>>(qhb, khb, vtb, ybf);
  // 5) output projection straight into d_out
  gemm_single_kernel<<<dim3(16, 16), 256, 0, stream>>>(ybf, wo_bf, out);
}
